// TransferLearningResNet34_1752346656976
// MI455X (gfx1250) — matrix operand order repaired and checked by reading, not compiled, not run on hardware
//
#include <hip/hip_runtime.h>
#include <hip/hip_bf16.h>
#include <math.h>

typedef __attribute__((ext_vector_type(16))) _Float16 v16h;
typedef __attribute__((ext_vector_type(8)))  _Float16 v8h;
typedef __attribute__((ext_vector_type(4)))  _Float16 v4h;
typedef __attribute__((ext_vector_type(8)))  float    v8f;

#define NB 16            // batch
#define EPSBN 1e-5f

// ---------------------------------------------------------------------------
// Implicit-GEMM conv + frozen-BN (+ optional residual, optional ReLU).
// GEMM: M = Cout (multiple of 64 in ResNet34), N = B*OH*OW (im2col gather),
// K = Cin*KH*KW. Workgroup: 256 threads = 8 waves; tile 64(M) x 32(N);
// K chunk = 64 (two v_wmma_f32_16x16x32_f16 per wave per barrier pair).
// Staging is branchless (clamped addresses + multiply-by-{0,1} masks), and
// the im2col (ic,r,s) decode is maintained INCREMENTALLY (k advances by a
// fixed 64 per chunk -> two-level carry update), so the inner loop contains
// no integer divisions.
// ---------------------------------------------------------------------------
__global__ __launch_bounds__(256)
void conv_bn_wmma(const float* __restrict__ x, const float* __restrict__ w,
                  const float* __restrict__ bg, const float* __restrict__ bb,
                  const float* __restrict__ bm, const float* __restrict__ bv,
                  const float* __restrict__ residual, float* __restrict__ out,
                  int Cin, int H, int W, int Cout,
                  int KH, int KW, int stride, int pad,
                  int OH, int OW, int relu)
{
    __shared__ _Float16 As[64][64];   // [m][k]
    __shared__ _Float16 Bs[32][64];   // [n][k]  (transposed im2col)

    const int K   = Cin * KH * KW;
    const int NP  = NB * OH * OW;
    const int ocB = blockIdx.x * 64;
    const int nB  = blockIdx.y * 32;

    const int tid  = threadIdx.x;
    const int lane = tid & 31;
    const int wave = tid >> 5;
    const int wm   = (wave >> 1) * 16;   // 0,16,32,48
    const int wn   = (wave & 1) * 16;    // 0,16

    // A-load mapping: 64 rows x 4 groups of 16 k (Cout % 64 == 0 -> row valid)
    const int am  = tid >> 2;
    const int ak0 = (tid & 3) * 16;
    const float* wrow = w + (long)(ocB + am) * K;

    // B-load mapping: 32 n x 8 groups of 8 k
    const int bn_ = tid >> 3;
    const int bk0 = (tid & 7) * 8;
    const int gn  = nB + bn_;
    const float gmask = (gn < NP) ? 1.0f : 0.0f;
    const int gnc = min(gn, NP - 1);
    const int ow0 = gnc % OW;
    const int t0_ = gnc / OW;
    const int oh0 = t0_ % OH;
    const int bi0 = t0_ / OH;
    const int KK  = KH * KW;

    // one-time decode of this thread's 8 k-slots (k = bk0 + j, all < 64 <= K)
    int icv[8], rv[8], sv[8];
    #pragma unroll
    for (int j = 0; j < 8; ++j) {
        int kk  = bk0 + j;
        int ic  = kk / KK;
        int rem = kk - ic * KK;
        int r   = rem / KW;
        int s   = rem - r * KW;
        icv[j] = ic; rv[j] = r; sv[j] = s;
    }
    // uniform per-chunk steps for k += 64 (two-level carry, each carry <= 1)
    const int dIC  = 64 / KK;
    const int dREM = 64 - dIC * KK;
    const int dR   = dREM / KW;
    const int dS   = dREM - dR * KW;

    const int  ihBase = oh0 * stride - pad;
    const int  iwBase = ow0 * stride - pad;
    const long xBase  = (long)bi0 * Cin * H * W;
    const int  HW     = H * W;

    v8f acc = {};

    for (int k0 = 0; k0 < K; k0 += 64) {
        // ---- stage A (weights): 16 halves/thread, 2x ds_store_b128 ----
        v8h a0, a1;
        #pragma unroll
        for (int j = 0; j < 8; ++j) {
            int   kk = k0 + ak0 + j;
            float v  = wrow[min(kk, K - 1)];
            a0[j] = (_Float16)(v * ((kk < K) ? 1.0f : 0.0f));
        }
        #pragma unroll
        for (int j = 0; j < 8; ++j) {
            int   kk = k0 + ak0 + 8 + j;
            float v  = wrow[min(kk, K - 1)];
            a1[j] = (_Float16)(v * ((kk < K) ? 1.0f : 0.0f));
        }
        *(v8h*)&As[am][ak0]     = a0;
        *(v8h*)&As[am][ak0 + 8] = a1;

        // ---- stage B (im2col gather): 8 halves/thread, 1x ds_store_b128 ----
        v8h b0;
        #pragma unroll
        for (int j = 0; j < 8; ++j) {
            int kk = k0 + bk0 + j;
            int ih = ihBase + rv[j];
            int iw = iwBase + sv[j];
            float msk = gmask *
                ((kk < K) ? 1.0f : 0.0f) *
                ((ih >= 0 && ih < H) ? 1.0f : 0.0f) *
                ((iw >= 0 && iw < W) ? 1.0f : 0.0f);
            int icc = min(icv[j], Cin - 1);      // K-tail: keep address valid
            int ihc = min(max(ih, 0), H - 1);
            int iwc = min(max(iw, 0), W - 1);
            float v = x[xBase + (long)icc * HW + ihc * W + iwc];
            b0[j] = (_Float16)(v * msk);

            // incremental (ic,r,s) += 64 with two-level carry
            int s2 = sv[j] + dS;
            int c1 = (s2 >= KW) ? 1 : 0;
            sv[j]  = s2 - (c1 ? KW : 0);
            int r2 = rv[j] + dR + c1;
            int c2 = (r2 >= KH) ? 1 : 0;
            rv[j]  = r2 - (c2 ? KH : 0);
            icv[j] += dIC + c2;
        }
        *(v8h*)&Bs[bn_][bk0] = b0;

        __syncthreads();

        // ---- two WMMAs per chunk; A and B both use contiguous k = 16*(lane/16) + i ----
        const int m   = lane & 15;
        const int kb  = (lane >> 4) * 8;
        const int n   = lane & 15;
        const int kb2 = (lane >> 4) * 16;
        #pragma unroll
        for (int kc = 0; kc < 64; kc += 32) {
            v16h af, bf;
            #pragma unroll
            for (int i = 0; i < 16; ++i)
                af[i] = As[wm + m][kc + kb2 + i];
            #pragma unroll
            for (int i = 0; i < 16; ++i)
                bf[i] = Bs[wn + n][kc + kb2 + i];
            acc = __builtin_amdgcn_wmma_f32_16x16x32_f16(false, af, false, bf,
                                                         (short)0, acc, false, false);
        }
        __syncthreads();
    }

    // ---- epilogue: BN fold, residual, ReLU, scatter to NCHW ----
    const int gcol = nB + wn + (lane & 15);
    if (gcol < NP) {
        int ow2 = gcol % OW; int t = gcol / OW; int oh2 = t % OH; int b2 = t / OH;
        #pragma unroll
        for (int r = 0; r < 8; ++r) {
            int oc = ocB + wm + (lane >> 4) * 8 + r;   // always < Cout
            float sc  = bg[oc] * rsqrtf(bv[oc] + EPSBN);
            float val = acc[r] * sc + bb[oc] - bm[oc] * sc;
            long  oi  = (((long)b2 * Cout + oc) * OH + oh2) * OW + ow2;
            if (residual) val += residual[oi];
            if (relu) val = fmaxf(val, 0.0f);
            out[oi] = val;
        }
    }
}

// ---------------------------------------------------------------------------
// Small-M FC WMMA GEMM: C[m][n] = sum_k A[m][k] * Wt[n][k] + bias[n]
// (optionally += existing C, optional ReLU). Tile: 16(M) x 128(N), K chunk 64.
// Call-site contract (holds for all uses): M == 16, K % 64 == 0, N % 128 == 0.
// ---------------------------------------------------------------------------
__global__ __launch_bounds__(256)
void fc_wmma(const float* __restrict__ A, int lda,
             const float* __restrict__ Wt, const float* __restrict__ bias,
             float* __restrict__ out, int ldo,
             int N, int K, int relu, int accumulate)
{
    __shared__ _Float16 As[16][64];
    __shared__ _Float16 Ws[128][64];

    const int nB   = blockIdx.x * 128;
    const int tid  = threadIdx.x;
    const int lane = tid & 31;
    const int wave = tid >> 5;
    const int wn   = wave * 16;

    v8f acc = {};

    // A staging map: m = tid>>4 (0..15), 4 k/thread (float4)
    const int sm  = tid >> 4;
    const int ska = (tid & 15) * 4;
    const float* arow = A + (long)sm * lda;

    // W staging map: n = tid>>1 (0..127), 32 k/thread (8x float4)
    const int sn  = tid >> 1;
    const int skw = (tid & 1) * 32;
    const float* wr = Wt + (long)(nB + sn) * K;

    for (int k0 = 0; k0 < K; k0 += 64) {
        // ---- stage A: one float4 load, one ds_store_b64 ----
        {
            float4 v = *(const float4*)(arow + k0 + ska);
            v4h tv = { (_Float16)v.x, (_Float16)v.y, (_Float16)v.z, (_Float16)v.w };
            *(v4h*)&As[sm][ska] = tv;
        }
        // ---- stage W: 8 float4 loads, 4x ds_store_b128 ----
        {
            const float4* w4 = (const float4*)(wr + k0 + skw);
            #pragma unroll
            for (int g = 0; g < 4; ++g) {
                float4 v0 = w4[2 * g];
                float4 v1 = w4[2 * g + 1];
                v8h t;
                t[0] = (_Float16)v0.x; t[1] = (_Float16)v0.y;
                t[2] = (_Float16)v0.z; t[3] = (_Float16)v0.w;
                t[4] = (_Float16)v1.x; t[5] = (_Float16)v1.y;
                t[6] = (_Float16)v1.z; t[7] = (_Float16)v1.w;
                *(v8h*)&Ws[sn][skw + 8 * g] = t;
            }
        }
        __syncthreads();

        const int m   = lane & 15;
        const int kb  = (lane >> 4) * 8;
        const int n   = lane & 15;
        const int kb2 = (lane >> 4) * 16;
        #pragma unroll
        for (int kc = 0; kc < 64; kc += 32) {
            v16h af, bf;
            #pragma unroll
            for (int i = 0; i < 16; ++i)
                af[i] = As[m][kc + kb2 + i];
            #pragma unroll
            for (int i = 0; i < 16; ++i)
                bf[i] = Ws[wn + n][kc + kb2 + i];
            acc = __builtin_amdgcn_wmma_f32_16x16x32_f16(false, af, false, bf,
                                                         (short)0, acc, false, false);
        }
        __syncthreads();
    }

    const int gcol = nB + wn + (lane & 15);   // always < N (N % 128 == 0)
    #pragma unroll
    for (int r = 0; r < 8; ++r) {
        int  row = (lane >> 4) * 8 + r;       // always < 16 == M
        long oi  = (long)row * ldo + gcol;
        float val = acc[r] + bias[gcol];
        if (accumulate) val += out[oi];
        if (relu) val = fmaxf(val, 0.0f);
        out[oi] = val;
    }
}

// ---------------------------------------------------------------------------
// Pointwise helpers
// ---------------------------------------------------------------------------
__global__ void maxpool3x3s2(const float* __restrict__ in, float* __restrict__ out,
                             int C, int H, int W, int OH, int OW)
{
    int tid = blockIdx.x * blockDim.x + threadIdx.x;
    int total = NB * C * OH * OW;
    if (tid >= total) return;
    int ow = tid % OW; int t = tid / OW;
    int oh = t % OH;   t /= OH;
    int c  = t % C;    int b = t / C;
    float m = -INFINITY;
    for (int r = 0; r < 3; ++r)
        for (int s = 0; s < 3; ++s) {
            int ih = oh * 2 - 1 + r, iw = ow * 2 - 1 + s;
            if (ih >= 0 && ih < H && iw >= 0 && iw < W)
                m = fmaxf(m, in[(((long)b * C + c) * H + ih) * W + iw]);
        }
    out[tid] = m;
}

__global__ void avgpool7x7(const float* __restrict__ in, float* __restrict__ out)
{
    int tid = blockIdx.x * blockDim.x + threadIdx.x;     // NB*512
    if (tid >= NB * 512) return;
    int c = tid % 512, b = tid / 512;
    const float* p = in + (((long)b * 512 + c) * 7) * 7;
    float s = 0.0f;
    #pragma unroll
    for (int i = 0; i < 49; ++i) s += p[i];
    out[tid] = s * (1.0f / 49.0f);
}

__global__ void embed_gather(const int* __restrict__ cap, const float* __restrict__ emb,
                             float* __restrict__ out)
{
    int tid = blockIdx.x * blockDim.x + threadIdx.x;     // NB*20*256
    if (tid >= NB * 20 * 256) return;
    int d = tid & 255;
    int t = (tid >> 8) % 20;
    int b = tid / (20 * 256);
    out[tid] = emb[(long)cap[b * 20 + t] * 256 + d];
}

__global__ void lstm_cell(const float* __restrict__ gates,
                          float* __restrict__ h, float* __restrict__ c)
{
    int tid = blockIdx.x * blockDim.x + threadIdx.x;     // NB*256
    if (tid >= NB * 256) return;
    int b = tid >> 8, j = tid & 255;
    const float* g = gates + (long)b * 1024;
    float ig = g[j], fg = g[256 + j], gg = g[512 + j], og = g[768 + j];
    float si = 1.0f / (1.0f + expf(-ig));
    float sf = 1.0f / (1.0f + expf(-fg));
    float so = 1.0f / (1.0f + expf(-og));
    float cc = sf * c[tid] + si * tanhf(gg);
    c[tid] = cc;
    h[tid] = so * tanhf(cc);
}

__global__ void zero_fill(float* __restrict__ p, int n)
{
    int tid = blockIdx.x * blockDim.x + threadIdx.x;
    if (tid < n) p[tid] = 0.0f;
}

// ---------------------------------------------------------------------------
// Host orchestration
// ---------------------------------------------------------------------------
struct BNP { const float *g, *b, *m, *v; };

extern "C" void kernel_launch(void* const* d_in, const int* in_sizes, int n_in,
                              void* d_out, int out_size, void* d_ws, size_t ws_size,
                              hipStream_t stream)
{
    (void)in_sizes; (void)n_in; (void)out_size; (void)ws_size;

    int idx = 0;
    auto F = [&](void) { return (const float*)d_in[idx++]; };

    const float* x        = F();                        // (16,3,224,224)
    const int*   captions = (const int*)d_in[idx++];    // (16,20)

    // ---- resnet_params walk (insertion order) ----
    const float* c1w = F();
    BNP bn1 = { F(), F(), F(), F() };

    const int cfg_oc[4] = { 64, 128, 256, 512 };
    const int cfg_nb[4] = { 3, 4, 6, 3 };
    const int cfg_ic[4] = { 64, 64, 128, 256 };

    const float* blk_w1[16];  BNP blk_b1[16];
    const float* blk_w2[16];  BNP blk_b2[16];
    const float* blk_wd[16];  BNP blk_bd[16];
    int nblk = 0;
    for (int li = 0; li < 4; ++li)
        for (int bi = 0; bi < cfg_nb[li]; ++bi) {
            blk_w1[nblk] = F(); blk_b1[nblk] = { F(), F(), F(), F() };
            blk_w2[nblk] = F(); blk_b2[nblk] = { F(), F(), F(), F() };
            if (bi == 0 && li > 0) {
                blk_wd[nblk] = F(); blk_bd[nblk] = { F(), F(), F(), F() };
            } else { blk_wd[nblk] = nullptr; blk_bd[nblk] = { 0, 0, 0, 0 }; }
            ++nblk;
        }

    const float* linear_w = F();  const float* linear_b = F();
    const float* embedding = F();
    const float* w_ih = F();  const float* w_hh = F();
    const float* b_ih = F();  const float* b_hh = F();
    const float* fc_w = F();  const float* fc_b = F();

    // ---- workspace layout ----
    char* wsb = (char*)d_ws;
    const size_t BIG_BYTES = (size_t)NB * 64 * 112 * 112 * 4;   // 51,380,224
    const size_t BUF_BYTES = (size_t)NB * 64 * 56 * 56 * 4;     // 12,845,056
    float* big = (float*)wsb;
    float* bb[4];
    bb[0] = (float*)(wsb + BIG_BYTES);          // outside big
    bb[1] = (float*)(wsb + 0);                  // alias big (dead after maxpool)
    bb[2] = (float*)(wsb + 13631488);
    bb[3] = (float*)(wsb + 27262976);
    size_t so = BIG_BYTES + BUF_BYTES;
    float* feats512 = (float*)(wsb + so); so += (size_t)NB * 512 * 4;
    float* featsD   = (float*)(wsb + so); so += (size_t)NB * 256 * 4;
    float* embAll   = (float*)(wsb + so); so += (size_t)NB * 20 * 256 * 4;
    float* hbuf     = (float*)(wsb + so); so += (size_t)NB * 256 * 4;
    float* cbuf     = (float*)(wsb + so); so += (size_t)NB * 256 * 4;
    float* gates    = (float*)(wsb + so); so += (size_t)NB * 1024 * 4;

    auto conv = [&](const float* in, const float* w, BNP bn, const float* res, float* out,
                    int Cin, int H, int W, int Cout, int KH, int KW,
                    int stride, int pad, int OH, int OW, int relu) {
        dim3 grid(Cout / 64, (NB * OH * OW + 31) / 32);
        conv_bn_wmma<<<grid, 256, 0, stream>>>(in, w, bn.g, bn.b, bn.m, bn.v,
                                               res, out, Cin, H, W, Cout,
                                               KH, KW, stride, pad, OH, OW, relu);
    };
    auto fc = [&](const float* A, int lda, const float* Wt, const float* bias,
                  float* out, int ldo, int N, int K, int relu, int accum) {
        fc_wmma<<<dim3(N / 128), 256, 0, stream>>>(A, lda, Wt, bias, out,
                                                   ldo, N, K, relu, accum);
    };

    // ---- stem: conv7x7 s2 + BN + ReLU, then maxpool 3x3 s2 ----
    conv(x, c1w, bn1, nullptr, big, 3, 224, 224, 64, 7, 7, 2, 3, 112, 112, 1);
    {
        int total = NB * 64 * 56 * 56;
        maxpool3x3s2<<<(total + 255) / 256, 256, 0, stream>>>(big, bb[0], 64, 112, 112, 56, 56);
    }

    // ---- residual blocks ----
    int cur = 0, Hc = 56, Wc = 56, k = 0;
    for (int li = 0; li < 4; ++li) {
        int oc = cfg_oc[li];
        for (int bi = 0; bi < cfg_nb[li]; ++bi, ++k) {
            int inC    = (bi == 0) ? cfg_ic[li] : oc;
            int stride = (li > 0 && bi == 0) ? 2 : 1;
            int OH = (Hc - 1) / stride + 1, OW = (Wc - 1) / stride + 1;
            int tmpI = (cur + 1) & 3, dsI = (cur + 2) & 3, outI = (cur + 3) & 3;

            conv(bb[cur], blk_w1[k], blk_b1[k], nullptr, bb[tmpI],
                 inC, Hc, Wc, oc, 3, 3, stride, 1, OH, OW, 1);

            const float* res;
            if (blk_wd[k]) {
                conv(bb[cur], blk_wd[k], blk_bd[k], nullptr, bb[dsI],
                     inC, Hc, Wc, oc, 1, 1, stride, 0, OH, OW, 0);
                res = bb[dsI];
            } else res = bb[cur];

            conv(bb[tmpI], blk_w2[k], blk_b2[k], res, bb[outI],
                 oc, OH, OW, oc, 3, 3, 1, 1, OH, OW, 1);

            cur = outI; Hc = OH; Wc = OW;
        }
    }

    // ---- head: avgpool -> linear(512->256) ----
    avgpool7x7<<<(NB * 512 + 255) / 256, 256, 0, stream>>>(bb[cur], feats512);
    fc(feats512, 512, linear_w, linear_b, featsD, 256, 256, 512, 0, 0);

    // ---- decoder ----
    embed_gather<<<(NB * 20 * 256 + 255) / 256, 256, 0, stream>>>(captions, embedding, embAll);
    zero_fill<<<(NB * 256 + 255) / 256, 256, 0, stream>>>(hbuf, NB * 256);
    zero_fill<<<(NB * 256 + 255) / 256, 256, 0, stream>>>(cbuf, NB * 256);

    const int V = 32000, T = 20;
    float* outp = (float*)d_out;                 // (B, T-1, V)

    // step 0: consumes image feature
    fc(featsD, 256, w_ih, b_ih, gates, 1024, 1024, 256, 0, 0);
    fc(hbuf,   256, w_hh, b_hh, gates, 1024, 1024, 256, 0, 1);
    lstm_cell<<<(NB * 256 + 255) / 256, 256, 0, stream>>>(gates, hbuf, cbuf);

    // steps 1..19: consume embeddings, emit logits
    for (int t = 1; t < T; ++t) {
        fc(embAll + (long)t * 256, T * 256, w_ih, b_ih, gates, 1024, 1024, 256, 0, 0);
        fc(hbuf, 256, w_hh, b_hh, gates, 1024, 1024, 256, 0, 1);
        lstm_cell<<<(NB * 256 + 255) / 256, 256, 0, stream>>>(gates, hbuf, cbuf);
        fc(hbuf, 256, fc_w, fc_b, outp + (long)(t - 1) * V, (long)(T - 1) * V,
           32000, 256, 0, 0);
    }
}
